// QuantumProjectionAttentionHybrid_57690000720067
// MI455X (gfx1250) — matrix operand order repaired and checked by reading, not compiled, not run on hardware
//
#include <hip/hip_runtime.h>

typedef __attribute__((ext_vector_type(16))) _Float16 v16h;
typedef __attribute__((ext_vector_type(8)))  _Float16 v8h;
typedef __attribute__((ext_vector_type(8)))  float    v8f;
typedef __attribute__((ext_vector_type(4)))  float    v4f;

#define Bc 8
#define Lc 1024
#define Sc 1024
#define Hc 8
#define Dc 64

// ---------------------------------------------------------------------------
// Stage 1: per-token angle projection -> 4-qubit circuit -> post projection.
// Emits f16. kmode=0: out[B,H,L,D] row-major (queries). kmode=1: out[B,H,D,S]
// (keys, pre-transposed so attention B-fragment loads are contiguous across lanes).
// ---------------------------------------------------------------------------
__global__ __launch_bounds__(256) void qc_proj_kernel(
    const float* __restrict__ xin,     // [B,L,H,D] tokens, token order (b,l,h)
    const float* __restrict__ pre_w,   // [64,4]
    const float* __restrict__ pre_b,   // [4]
    const float* __restrict__ qcw,     // [1,4,3]
    const float* __restrict__ post_w,  // [68,64]
    const float* __restrict__ post_b,  // [64]
    _Float16* __restrict__ out,
    int kmode)
{
  __shared__ float sW[68 * 64];
  __shared__ float sBv[64];
  __shared__ float sPre[64 * 4];
  __shared__ float sPreB[4];
  __shared__ float sRot[4][8];

  for (int i = threadIdx.x; i < 68 * 64; i += 256) sW[i] = post_w[i];
  if (threadIdx.x < 64)  sBv[threadIdx.x]  = post_b[threadIdx.x];
  sPre[threadIdx.x] = pre_w[threadIdx.x];           // exactly 256 elements
  if (threadIdx.x < 4)   sPreB[threadIdx.x] = pre_b[threadIdx.x];
  if (threadIdx.x < 4) {
    int w = threadIdx.x;
    float phi = qcw[w * 3 + 0], th = qcw[w * 3 + 1], om = qcw[w * 3 + 2];
    float c = cosf(0.5f * th), s = sinf(0.5f * th);
    float apo = 0.5f * (phi + om), amo = 0.5f * (phi - om);
    // Rot = [[e^{-i apo} c, -e^{+i amo} s],[e^{-i amo} s, e^{+i apo} c]]
    sRot[w][0] =  cosf(apo) * c;  sRot[w][1] = -sinf(apo) * c;   // u00
    sRot[w][2] = -cosf(amo) * s;  sRot[w][3] = -sinf(amo) * s;   // u01
    sRot[w][4] =  cosf(amo) * s;  sRot[w][5] = -sinf(amo) * s;   // u10
    sRot[w][6] =  cosf(apo) * c;  sRot[w][7] =  sinf(apo) * c;   // u11
  }
  __syncthreads();

  int t = blockIdx.x * 256 + threadIdx.x;   // token id, order (b,l,h)
  if (t >= Bc * Lc * Hc) return;

  const float* xp = xin + (size_t)t * Dc;
  float x[64];
#pragma unroll
  for (int k = 0; k < 64; k += 4) {
    v4f v = *(const v4f*)(xp + k);
    x[k] = v.x; x[k + 1] = v.y; x[k + 2] = v.z; x[k + 3] = v.w;
  }

  // angles: tanh(x @ pre_w + pre_b) * (pi/2)  (half-angle of RX(a*pi))
  float half_ang[4];
#pragma unroll
  for (int j = 0; j < 4; j++) {
    float a = sPreB[j];
    for (int k = 0; k < 64; k++) a += x[k] * sPre[k * 4 + j];
    half_ang[j] = tanhf(a) * 1.57079632679489662f;
  }

  // 4-qubit state |0000>, wire w maps to bit (3-w) of the flat index
  float re[16], im[16];
#pragma unroll
  for (int i = 0; i < 16; i++) { re[i] = 0.f; im[i] = 0.f; }
  re[0] = 1.f;

  // RX(2*half_ang[w]) on each wire: U = [[c, -i s],[-i s, c]]
#pragma unroll
  for (int w = 0; w < 4; w++) {
    float c = cosf(half_ang[w]), s = sinf(half_ang[w]);
    int st = 8 >> w;
#pragma unroll
    for (int i = 0; i < 16; i++) {
      if (i & st) continue;
      int j2 = i + st;
      float r0 = re[i], i0 = im[i], r1 = re[j2], i1 = im[j2];
      re[i]  =  c * r0 + s * i1;   im[i]  =  c * i0 - s * r1;
      re[j2] =  s * i0 + c * r1;   im[j2] = -s * r0 + c * i1;
    }
  }
  // one layer of shared Rot gates
#pragma unroll
  for (int w = 0; w < 4; w++) {
    float u00r = sRot[w][0], u00i = sRot[w][1], u01r = sRot[w][2], u01i = sRot[w][3];
    float u10r = sRot[w][4], u10i = sRot[w][5], u11r = sRot[w][6], u11i = sRot[w][7];
    int st = 8 >> w;
#pragma unroll
    for (int i = 0; i < 16; i++) {
      if (i & st) continue;
      int j2 = i + st;
      float r0 = re[i], i0 = im[i], r1 = re[j2], i1 = im[j2];
      re[i]  = u00r * r0 - u00i * i0 + u01r * r1 - u01i * i1;
      im[i]  = u00r * i0 + u00i * r0 + u01r * i1 + u01i * r1;
      re[j2] = u10r * r0 - u10i * i0 + u11r * r1 - u11i * i1;
      im[j2] = u10r * i0 + u10i * r0 + u11r * i1 + u11i * r1;
    }
  }
  // ring CNOTs (0,1)(1,2)(2,3)(3,0): where control bit set, flip target bit
#pragma unroll
  for (int w = 0; w < 4; w++) {
    int cb = 8 >> w;
    int tb = 8 >> ((w + 1) & 3);
#pragma unroll
    for (int i = 0; i < 16; i++) {
      if ((i & cb) && !(i & tb)) {
        int j2 = i | tb;
        float tr = re[i], ti = im[i];
        re[i] = re[j2]; im[i] = im[j2];
        re[j2] = tr;    im[j2] = ti;
      }
    }
  }
  // <Z_w> expectations
  float z[4];
#pragma unroll
  for (int w = 0; w < 4; w++) {
    int bm = 8 >> w;
    float acc = 0.f;
#pragma unroll
    for (int i = 0; i < 16; i++) {
      float p = re[i] * re[i] + im[i] * im[i];
      acc += (i & bm) ? -p : p;
    }
    z[w] = acc;
  }

  // post projection: concat(x, z) @ post_w + post_b  -> f16 out
  int h  = t % Hc;
  int bl = t / Hc;
  int l  = bl % Lc;
  int b  = bl / Lc;
  int bh = b * Hc + h;

  if (!kmode) {
    _Float16* op = out + ((size_t)bh * Lc + l) * Dc;   // [B,H,L,D]
    for (int d8 = 0; d8 < 64; d8 += 8) {
      v8h pk;
#pragma unroll
      for (int jj = 0; jj < 8; jj++) {
        int d = d8 + jj;
        float acc = sBv[d];
        for (int k = 0; k < 64; k++) acc += x[k] * sW[k * 64 + d];
#pragma unroll
        for (int j = 0; j < 4; j++) acc += z[j] * sW[(64 + j) * 64 + d];
        pk[jj] = (_Float16)acc;
      }
      *(v8h*)(op + d8) = pk;
    }
  } else {
    _Float16* op = out + (size_t)bh * Dc * Sc;         // [B,H,D,S]
    for (int d = 0; d < 64; d++) {
      float acc = sBv[d];
      for (int k = 0; k < 64; k++) acc += x[k] * sW[k * 64 + d];
#pragma unroll
      for (int j = 0; j < 4; j++) acc += z[j] * sW[(64 + j) * 64 + d];
      op[(size_t)d * Sc + l] = (_Float16)acc;
    }
  }
}

// ---------------------------------------------------------------------------
// Stage 2: attention. One wave (32 threads) per 16 query rows of one (b,h).
// scores -> LDS (16x1024 f32) -> softmax -> attn out + probs -> context WMMA.
// ---------------------------------------------------------------------------
__global__ __launch_bounds__(32) void attn_kernel(
    const _Float16* __restrict__ qp,   // [B,H,L,D]
    const _Float16* __restrict__ kpt,  // [B,H,D,S]  (K^T)
    const float*    __restrict__ vin,  // [B,S,H,D] f32
    float* __restrict__ attn_out,      // [B,H,L,S]
    float* __restrict__ ctx_out)       // [B,L,H,D]
{
  extern __shared__ float sP[];        // 16 * Sc floats = 64KB

  int wg = blockIdx.x;
  int mt = wg % (Lc / 16);
  int bh = wg / (Lc / 16);
  int h  = bh % Hc;
  int b  = bh / Hc;
  int m0 = mt * 16;

  int lane    = threadIdx.x;
  int mrow    = lane & 15;
  int halfsel = lane >> 4;

  // ---- A fragments for Qp rows m0..m0+15 (K = D split into 32+32) ----
  const _Float16* qbase = qp + ((size_t)bh * Lc + m0) * Dc;
  v16h a0, a1;
  {
    const _Float16* qrow = qbase + (size_t)mrow * Dc;
    v8h lo0 = *(const v8h*)(qrow +  0 + 8 * halfsel);
    v8h hi0 = *(const v8h*)(qrow + 16 + 8 * halfsel);
    v8h lo1 = *(const v8h*)(qrow + 32 + 8 * halfsel);
    v8h hi1 = *(const v8h*)(qrow + 48 + 8 * halfsel);
#pragma unroll
    for (int j = 0; j < 8; j++) {
      a0[j] = lo0[j]; a0[j + 8] = hi0[j];
      a1[j] = lo1[j]; a1[j + 8] = hi1[j];
    }
  }

  // ---- scores: for each 16-wide S tile, 2 WMMA (kk=0,32), store to LDS ----
  const _Float16* ktbase = kpt + (size_t)bh * Dc * Sc;
  const float scale = 0.125f;   // 1/sqrt(64)
  for (int nt = 0; nt < Sc / 16; nt++) {
    int s0 = nt * 16;
    // B fragment: lane = N column (s), elements = K (d) in table order
    v16h b0, b1;
#pragma unroll
    for (int e = 0; e < 16; e++) {
      int kd = (e < 8) ? (8 * halfsel + e) : (8 + 8 * halfsel + e);
      b0[e] = ktbase[(size_t)kd        * Sc + s0 + mrow];
      b1[e] = ktbase[(size_t)(32 + kd) * Sc + s0 + mrow];
    }
    v8f c = {};
    c = __builtin_amdgcn_wmma_f32_16x16x32_f16(false, a0, false, b0, (short)0, c, false, false);
    c = __builtin_amdgcn_wmma_f32_16x16x32_f16(false, a1, false, b1, (short)0, c, false, false);
#pragma unroll
    for (int r = 0; r < 8; r++)
      sP[(size_t)(r + 8 * halfsel) * Sc + s0 + mrow] = c[r] * scale;
  }
  __syncthreads();

  // ---- softmax per row; write normalized probs to attn output and LDS ----
  float* arow_base = attn_out + ((size_t)bh * Lc + m0) * Sc;
  for (int r = 0; r < 16; r++) {
    float* row = sP + (size_t)r * Sc;
    float mx = -3.402823466e38f;
    for (int i = lane; i < Sc; i += 32) mx = fmaxf(mx, row[i]);
#pragma unroll
    for (int off = 16; off; off >>= 1) mx = fmaxf(mx, __shfl_xor(mx, off, 32));
    float sum = 0.f;
    for (int i = lane; i < Sc; i += 32) {
      float e = __expf(row[i] - mx);
      row[i] = e;
      sum += e;
    }
#pragma unroll
    for (int off = 16; off; off >>= 1) sum += __shfl_xor(sum, off, 32);
    float inv = 1.0f / sum;
    float* arow = arow_base + (size_t)r * Sc;
    for (int i = lane; i < Sc; i += 32) {
      float p = row[i] * inv;
      row[i] = p;
      arow[i] = p;
    }
  }
  __syncthreads();

  // ---- context: P[16,1024] @ V[1024,64], K tiles of 32 over S ----
  v8f acc[4] = {};
  const float* vbase = vin + ((size_t)b * Sc) * (Hc * Dc) + (size_t)h * Dc;
  for (int kt = 0; kt < Sc / 32; kt++) {
    int sb = kt * 32;
    // A fragment from LDS probs (cvt f32->f16)
    v16h a;
#pragma unroll
    for (int j = 0; j < 8; j++) {
      a[j]     = (_Float16)sP[(size_t)mrow * Sc + sb +       8 * halfsel + j];
      a[j + 8] = (_Float16)sP[(size_t)mrow * Sc + sb + 16 + 8 * halfsel + j];
    }
    // B fragment: lane = N column (d), elements = K (s) in table order
    const float* vrow = vbase + (size_t)sb * (Hc * Dc) + mrow;
#pragma unroll
    for (int dt = 0; dt < 4; dt++) {
      int d0 = dt * 16;
      v16h bb;
#pragma unroll
      for (int e = 0; e < 16; e++) bb[e] = (_Float16)vrow[(size_t)((e < 8) ? (8 * halfsel + e) : (8 + 8 * halfsel + e)) * (Hc * Dc) + d0];
      acc[dt] = __builtin_amdgcn_wmma_f32_16x16x32_f16(false, a, false, bb, (short)0, acc[dt], false, false);
    }
  }

  // ---- write context [B,L,H,D] ----
#pragma unroll
  for (int dt = 0; dt < 4; dt++) {
#pragma unroll
    for (int r = 0; r < 8; r++) {
      int mr = r + 8 * halfsel;
      ctx_out[(((size_t)b * Lc + m0 + mr) * Hc + h) * Dc + dt * 16 + mrow] = acc[dt][r];
    }
  }
}

// ---------------------------------------------------------------------------
extern "C" void kernel_launch(void* const* d_in, const int* in_sizes, int n_in,
                              void* d_out, int out_size, void* d_ws, size_t ws_size,
                              hipStream_t stream) {
  (void)in_sizes; (void)n_in; (void)out_size; (void)ws_size;
  const float* queries  = (const float*)d_in[0];
  const float* keys     = (const float*)d_in[1];
  const float* values   = (const float*)d_in[2];
  const float* pre_q_w  = (const float*)d_in[3];
  const float* pre_q_b  = (const float*)d_in[4];
  const float* pre_k_w  = (const float*)d_in[5];
  const float* pre_k_b  = (const float*)d_in[6];
  const float* qc_w     = (const float*)d_in[7];
  const float* kc_w     = (const float*)d_in[8];
  const float* post_q_w = (const float*)d_in[9];
  const float* post_q_b = (const float*)d_in[10];
  const float* post_k_w = (const float*)d_in[11];
  const float* post_k_b = (const float*)d_in[12];

  float* ctx  = (float*)d_out;                            // [B,L,H,D]
  float* attn = ctx + (size_t)Bc * Lc * Hc * Dc;          // [B,H,L,S]

  _Float16* qp  = (_Float16*)d_ws;                        // [B,H,L,D] f16
  _Float16* kpt = qp + (size_t)Bc * Hc * Lc * Dc;         // [B,H,D,S] f16

  int ntok = Bc * Lc * Hc;   // 65536
  qc_proj_kernel<<<ntok / 256, 256, 0, stream>>>(
      queries, pre_q_w, pre_q_b, qc_w, post_q_w, post_q_b, qp, 0);
  qc_proj_kernel<<<ntok / 256, 256, 0, stream>>>(
      keys, pre_k_w, pre_k_b, kc_w, post_k_w, post_k_b, kpt, 1);

  dim3 grid(Bc * Hc * (Lc / 16));   // 4096 waves
  attn_kernel<<<grid, 32, 16 * Sc * sizeof(float), stream>>>(
      qp, kpt, values, attn, ctx);
}
